// RNAModel_7249904796459
// MI455X (gfx1250) — compile-verified, matrix operand order checked by reading, not run on hardware
//
#include <hip/hip_runtime.h>
#include <hip/hip_bf16.h>
#include <math.h>

// ---------------------------------------------------------------------------
// Model dims
#define Bsz 32
#define Ssz 512
#define Dsz 768
#define Hh  12
#define HD  64
#define Ll  12
#define DFF 3072
#define ROWS (Bsz * Ssz)        // 16384
#define EPSF 1e-5f

typedef __attribute__((ext_vector_type(8)))  float        v8f;
typedef __attribute__((ext_vector_type(16))) __bf16       v16bf;
typedef __attribute__((ext_vector_type(4)))  unsigned int v4u;
typedef __attribute__((ext_vector_type(8)))  int          v8i;
typedef __attribute__((ext_vector_type(4)))  int          v4i;

static __device__ inline v8f v8f_zero() {
    v8f z;
#pragma unroll
    for (int i = 0; i < 8; ++i) z[i] = 0.f;
    return z;
}

static __device__ inline float gelu_exact(float x) {
    return 0.5f * x * (1.0f + erff(x * 0.70710678118654752440f));
}

static __device__ inline unsigned ldsoff(const void* p) {
    return (unsigned)(unsigned long long)p;   // low 32 bits of flat LDS addr = LDS byte offset
}

// --- CDNA5 async global->LDS copy (ASYNCcnt) --------------------------------
#define ASYNC_B128(ldsptr, gptr)                                                   \
    asm volatile("global_load_async_to_lds_b128 %0, %1, off"                       \
                 :: "v"(ldsoff((const void*)(ldsptr))), "v"((const void*)(gptr))   \
                 : "memory")
#define WAIT_ASYNC0 asm volatile("s_wait_asynccnt 0" ::: "memory")

// --- CDNA5 Tensor Data Mover: 2D tile (rows x 32 bf16) global -> LDS --------
// Descriptor packing per CDNA5 ISA.
// tile_dim0 = 32 elements (contiguous), tile_dim1 = tile_rows,
// tensor_dim1 = valid_rows (rows beyond read as zero), stride = row stride (elems).
// This toolchain exposes the 6-arg builtin: (v4u g0, v8i g1, v4i g2, v4i g3, v8i, i32 cpol)
static __device__ inline void tdm_load_tile(const void* gaddr, unsigned lds_addr,
                                            unsigned tile_rows, unsigned valid_rows,
                                            unsigned stride_elems) {
    unsigned long long ga = (unsigned long long)gaddr;
    v4u g0;
    g0[0] = 1u;                                            // count=1, user mode
    g0[1] = lds_addr;                                      // lds_addr [63:32]
    g0[2] = (unsigned)ga;                                  // global_addr lo
    g0[3] = (unsigned)((ga >> 32) & 0x01FFFFFFu) | 0x80000000u; // addr[56:32] | type=2
    v8i g1;
    g1[0] = (int)(1u << 16);                               // workgroup_mask=0, data_size=1 (2B)
    g1[1] = (int)(32u << 16);                              // tensor_dim0 lo16 = 32
    g1[2] = (int)((valid_rows & 0xFFFFu) << 16);           // tensor_dim0 hi=0 | tensor_dim1 lo16
    g1[3] = (int)(32u << 16);                              // tensor_dim1 hi=0 | tile_dim0=32
    g1[4] = (int)(tile_rows & 0xFFFFu);                    // tile_dim1 | tile_dim2=0
    g1[5] = (int)stride_elems;                             // tensor_dim0_stride lo32
    g1[6] = 0;                                             // stride hi16 | dim1_stride lo16
    g1[7] = 0;                                             // dim1_stride hi
    v4i g2 = {0, 0, 0, 0};
    v4i g3 = {0, 0, 0, 0};
    v8i g4 = {0, 0, 0, 0, 0, 0, 0, 0};
    __builtin_amdgcn_tensor_load_to_lds(g0, g1, g2, g3, g4, 0);
}

// ---------------------------------------------------------------------------
// Embedding: x[b,s,:] = tok_emb[tok[b,s]] + pos_emb[s]  (fp32)
__global__ __launch_bounds__(256)
void embed_kernel(const int* __restrict__ tok, const float* __restrict__ te,
                  const float* __restrict__ pe, float* __restrict__ x) {
    size_t i = (size_t)blockIdx.x * 256 + threadIdx.x;
    if (i >= (size_t)ROWS * Dsz) return;
    int bs = (int)(i / Dsz);
    int d  = (int)(i % Dsz);
    int s  = bs % Ssz;
    int t  = tok[bs];
    x[i] = te[(size_t)t * Dsz + d] + pe[(size_t)s * Dsz + d];
}

// ---------------------------------------------------------------------------
// fp32 [K][N]  ->  bf16 [N][K]  (transpose + downconvert), 32x32 LDS tiles
__global__ __launch_bounds__(256)
void transpose_bf16_kernel(const float* __restrict__ src, __hip_bfloat16* __restrict__ dst,
                           int K, int N) {
    __shared__ float t[32][33];
    const int tx = threadIdx.x & 31, ty = threadIdx.x >> 5;  // 32 x 8
    const int kb = blockIdx.y * 32, nb = blockIdx.x * 32;
#pragma unroll
    for (int i = 0; i < 4; ++i) {
        int k = kb + ty + i * 8, n = nb + tx;
        t[ty + i * 8][tx] = (k < K && n < N) ? src[(size_t)k * N + n] : 0.f;
    }
    __syncthreads();
#pragma unroll
    for (int i = 0; i < 4; ++i) {
        int n = nb + ty + i * 8, k = kb + tx;
        if (n < N && k < K) dst[(size_t)n * K + k] = __float2bfloat16(t[tx][ty + i * 8]);
    }
}

// ---------------------------------------------------------------------------
// LayerNorm over D=768: bf16 out (always) + optional fp32 out
__global__ __launch_bounds__(256)
void layernorm_kernel(const float* __restrict__ in, const float* __restrict__ sc,
                      const float* __restrict__ bi, __hip_bfloat16* __restrict__ out_bf,
                      float* __restrict__ out_f32) {
    __shared__ float rs[8], rq[8];
    const int row  = blockIdx.x;
    const int tid  = threadIdx.x;
    const int lane = tid & 31;
    const int w    = tid >> 5;
    const float* x = in + (size_t)row * Dsz;

    float vals[3];
    float s = 0.f, q = 0.f;
#pragma unroll
    for (int i = 0; i < 3; ++i) {
        float v = x[tid + i * 256];
        vals[i] = v;
        s += v;
        q += v * v;
    }
#pragma unroll
    for (int off = 16; off > 0; off >>= 1) {
        s += __shfl_xor(s, off, 32);
        q += __shfl_xor(q, off, 32);
    }
    if (lane == 0) { rs[w] = s; rq[w] = q; }
    __syncthreads();
    s = 0.f; q = 0.f;
#pragma unroll
    for (int i = 0; i < 8; ++i) { s += rs[i]; q += rq[i]; }
    const float mean = s * (1.0f / Dsz);
    const float var  = q * (1.0f / Dsz) - mean * mean;
    const float rstd = rsqrtf(var + EPSF);
#pragma unroll
    for (int i = 0; i < 3; ++i) {
        int c = tid + i * 256;
        float y = (vals[i] - mean) * rstd * sc[c] + bi[c];
        out_bf[(size_t)row * Dsz + c] = __float2bfloat16(y);
        if (out_f32) out_f32[(size_t)row * Dsz + c] = y;
    }
}

// ---------------------------------------------------------------------------
// TDM-fed, double-buffered bf16 WMMA GEMM:
//   C[M,N] = act( A[M,K] @ Bt[N,K]^T + bias + resid )
// Block: 256 thr = 8 waves; tile 128(M) x 64(N); K-step 32.
// Wave0 drives the Tensor Data Mover; everyone else just computes.
#define TM 128
#define TN 64
#define TK 32

template <bool GELU, bool OUTBF>
__global__ __launch_bounds__(256)
void gemm_bf16_wmma(const __hip_bfloat16* __restrict__ A,
                    const __hip_bfloat16* __restrict__ Bt,
                    const float* __restrict__ bias, const float* __restrict__ resid,
                    void* __restrict__ Cout, int M, int K, int N) {
    __shared__ __hip_bfloat16 As[2][TM][TK];
    __shared__ __hip_bfloat16 Bs[2][TN][TK];

    const int tid  = threadIdx.x;
    const int lane = tid & 31;
    const int w    = tid >> 5;
    const int wr   = w & 3;       // wave M offset = 32*wr
    const int wc   = w >> 2;      // wave N offset = 32*wc
    const int half = lane >> 4;
    const int l16  = lane & 15;
    const int mblk = blockIdx.y * TM;
    const int nblk = blockIdx.x * TN;
    int nvalid = N - nblk; if (nvalid > TN) nvalid = TN;

    v8f acc[2][2];
#pragma unroll
    for (int i = 0; i < 2; ++i)
#pragma unroll
        for (int j = 0; j < 2; ++j) acc[i][j] = v8f_zero();

#define GEMM_ISSUE(bufi, kbv)                                                           \
    do {                                                                                \
        tdm_load_tile(A  + (size_t)mblk * K + (kbv), ldsoff(&As[bufi][0][0]),           \
                      TM, TM, (unsigned)K);                                             \
        tdm_load_tile(Bt + (size_t)nblk * K + (kbv), ldsoff(&Bs[bufi][0][0]),           \
                      TN, (unsigned)nvalid, (unsigned)K);                               \
    } while (0)

    if (w == 0) GEMM_ISSUE(0, 0);

    int buf = 0;
    for (int kb = 0; kb < K; kb += TK) {
        const bool more = (kb + TK) < K;
        if (w == 0) {
            if (more) {
                GEMM_ISSUE(buf ^ 1, kb + TK);
                __builtin_amdgcn_s_wait_tensorcnt(2);   // current tile's 2 DMAs done
            } else {
                __builtin_amdgcn_s_wait_tensorcnt(0);
            }
        }
        asm volatile("" ::: "memory");
        __syncthreads();

        // ---- fragments (two contiguous 16B ds loads each)
        v16bf a[2], b[2];
#pragma unroll
        for (int i = 0; i < 2; ++i) {
            int m = wr * 32 + i * 16 + l16;
            const __bf16* ap = reinterpret_cast<const __bf16*>(&As[buf][m][half * 8]);
#pragma unroll
            for (int t = 0; t < 8; ++t) { a[i][t] = ap[t]; a[i][8 + t] = ap[16 + t]; }
        }
#pragma unroll
        for (int j = 0; j < 2; ++j) {
            int n = wc * 32 + j * 16 + l16;
            const __bf16* bp = reinterpret_cast<const __bf16*>(&Bs[buf][n][half * 8]);
#pragma unroll
            for (int t = 0; t < 8; ++t) { b[j][t] = bp[t]; b[j][8 + t] = bp[16 + t]; }
        }
#pragma unroll
        for (int i = 0; i < 2; ++i)
#pragma unroll
            for (int j = 0; j < 2; ++j)
                acc[i][j] = __builtin_amdgcn_wmma_f32_16x16x32_bf16(
                    false, a[i], false, b[j], (short)0, acc[i][j], false, false);

        __syncthreads();
        buf ^= 1;
    }
#undef GEMM_ISSUE

    // ---- epilogue
#pragma unroll
    for (int i = 0; i < 2; ++i)
#pragma unroll
        for (int j = 0; j < 2; ++j)
#pragma unroll
            for (int p = 0; p < 8; ++p) {
                int m = mblk + wr * 32 + i * 16 + half * 8 + p;
                int n = nblk + wc * 32 + j * 16 + l16;
                if (m < M && n < N) {
                    float v = acc[i][j][p];
                    if (bias)  v += bias[n];
                    if (resid) v += resid[(size_t)m * N + n];
                    if (GELU)  v = gelu_exact(v);
                    if (OUTBF) ((__hip_bfloat16*)Cout)[(size_t)m * N + n] = __float2bfloat16(v);
                    else       ((float*)Cout)[(size_t)m * N + n] = v;
                }
            }
}

// ---------------------------------------------------------------------------
// Flash attention (bf16 in/out): block = (128-query tile, head, batch).
// Q/K tiles arrive via global_load_async_to_lds_b128; online softmax via
// 16-lane shfl_xor; P re-laid to A-fragment format through per-wave LDS.
__global__ __launch_bounds__(256)
void attn_kernel(const __hip_bfloat16* __restrict__ Qg, const __hip_bfloat16* __restrict__ Kg,
                 const __hip_bfloat16* __restrict__ Vg, __hip_bfloat16* __restrict__ Og) {
    __shared__ __hip_bfloat16 Qs[128][64];
    __shared__ __hip_bfloat16 Ks[32][64];
    __shared__ __hip_bfloat16 Vs[64][32];   // transposed: [d][key]
    __shared__ __hip_bfloat16 Ps[8][16][32];

    const int tid  = threadIdx.x;
    const int lane = tid & 31;
    const int w    = tid >> 5;
    const int half = lane >> 4;
    const int l16  = lane & 15;
    const int qb   = blockIdx.x * 128;
    const int h    = blockIdx.y;
    const int b    = blockIdx.z;

    const __hip_bfloat16* qbase = Qg + (size_t)b * Ssz * Dsz + (size_t)h * HD;
    const __hip_bfloat16* kbase = Kg + (size_t)b * Ssz * Dsz + (size_t)h * HD;
    const __hip_bfloat16* vbase = Vg + (size_t)b * Ssz * Dsz + (size_t)h * HD;

    // stage Q tile once (128x64 bf16 = 1024 x 16B chunks, async DMA)
#pragma unroll
    for (int u = 0; u < 4; ++u) {
        int c = tid + 256 * u;
        int row = c >> 3, sub = c & 7;
        ASYNC_B128(&Qs[row][sub * 8], qbase + (size_t)(qb + row) * Dsz + sub * 8);
    }
    WAIT_ASYNC0;
    __syncthreads();

    float rowm[8], rowl[8];
    v8f cacc[4];
#pragma unroll
    for (int p = 0; p < 8; ++p) { rowm[p] = -1e30f; rowl[p] = 0.f; }
#pragma unroll
    for (int j = 0; j < 4; ++j) cacc[j] = v8f_zero();

    const float scl = 0.125f;   // 1/sqrt(64)

    for (int kb = 0; kb < Ssz; kb += 32) {
        // K block 32x64: async DMA, one 16B chunk per thread
        {
            int row = tid >> 3, sub = tid & 7;
            ASYNC_B128(&Ks[row][sub * 8], kbase + (size_t)(kb + row) * Dsz + sub * 8);
        }
        // V block transposed: plain bf16 copy (no conversion)
        {
            int d = tid >> 2, k0 = (tid & 3) * 8;
#pragma unroll
            for (int i = 0; i < 8; ++i)
                Vs[d][k0 + i] = vbase[(size_t)(kb + k0 + i) * Dsz + d];
        }
        WAIT_ASYNC0;
        __syncthreads();

        // scores: 16(q) x 32(key) per wave, contract over HD=64 in two steps
        v8f s0 = v8f_zero(), s1 = v8f_zero();
#pragma unroll
        for (int ks = 0; ks < 64; ks += 32) {
            v16bf aq;
            {
                int m = w * 16 + l16;
                const __bf16* ap = reinterpret_cast<const __bf16*>(&Qs[m][ks + half * 8]);
#pragma unroll
                for (int t = 0; t < 8; ++t) { aq[t] = ap[t]; aq[8 + t] = ap[16 + t]; }
            }
            v16bf bk0, bk1;
            {
                const __bf16* bp = reinterpret_cast<const __bf16*>(&Ks[l16][ks + half * 8]);
#pragma unroll
                for (int t = 0; t < 8; ++t) { bk0[t] = bp[t]; bk0[8 + t] = bp[16 + t]; }
            }
            {
                const __bf16* bp = reinterpret_cast<const __bf16*>(&Ks[16 + l16][ks + half * 8]);
#pragma unroll
                for (int t = 0; t < 8; ++t) { bk1[t] = bp[t]; bk1[8 + t] = bp[16 + t]; }
            }
            s0 = __builtin_amdgcn_wmma_f32_16x16x32_bf16(false, aq, false, bk0, (short)0, s0, false, false);
            s1 = __builtin_amdgcn_wmma_f32_16x16x32_bf16(false, aq, false, bk1, (short)0, s1, false, false);
        }

        // online softmax per query row (row lives across 16 lanes of a half)
#pragma unroll
        for (int p = 0; p < 8; ++p) {
            float v0 = s0[p] * scl, v1 = s1[p] * scl;
            float mx = fmaxf(v0, v1);
#pragma unroll
            for (int off = 8; off > 0; off >>= 1) mx = fmaxf(mx, __shfl_xor(mx, off, 32));
            float mn    = fmaxf(rowm[p], mx);
            float alpha = __expf(rowm[p] - mn);
            float p0 = __expf(v0 - mn);
            float p1 = __expf(v1 - mn);
            float ls = p0 + p1;
#pragma unroll
            for (int off = 8; off > 0; off >>= 1) ls += __shfl_xor(ls, off, 32);
            rowl[p] = rowl[p] * alpha + ls;
            rowm[p] = mn;
#pragma unroll
            for (int j = 0; j < 4; ++j) cacc[j][p] *= alpha;
            int rr = half * 8 + p;
            Ps[w][rr][l16]      = __float2bfloat16(p0);
            Ps[w][rr][16 + l16] = __float2bfloat16(p1);
        }
        asm volatile("s_wait_dscnt 0" ::: "memory");  // intra-wave LDS RAW

        // P(16x32) @ V(32x64): 4 WMMAs
        v16bf apv;
        {
            const __bf16* pp = reinterpret_cast<const __bf16*>(&Ps[w][l16][half * 8]);
#pragma unroll
            for (int t = 0; t < 8; ++t) { apv[t] = pp[t]; apv[8 + t] = pp[16 + t]; }
        }
#pragma unroll
        for (int j = 0; j < 4; ++j) {
            v16bf bv;
            int d = j * 16 + l16;
            const __bf16* vp = reinterpret_cast<const __bf16*>(&Vs[d][half * 8]);
#pragma unroll
            for (int t = 0; t < 8; ++t) { bv[t] = vp[t]; bv[8 + t] = vp[16 + t]; }
            cacc[j] = __builtin_amdgcn_wmma_f32_16x16x32_bf16(false, apv, false, bv, (short)0, cacc[j], false, false);
        }
        __syncthreads();
    }

    // write context (normalized, bf16)
#pragma unroll
    for (int j = 0; j < 4; ++j)
#pragma unroll
        for (int p = 0; p < 8; ++p) {
            int m = qb + w * 16 + half * 8 + p;
            int d = j * 16 + l16;
            Og[((size_t)b * Ssz + m) * Dsz + h * HD + d] = __float2bfloat16(cacc[j][p] / rowl[p]);
        }
}

// ---------------------------------------------------------------------------
__global__ __launch_bounds__(256)
void selected_kernel(const unsigned char* __restrict__ sel, float* __restrict__ out, int n) {
    int i = blockIdx.x * 256 + threadIdx.x;
    if (i < n) out[i] = sel[i] ? 1.0f : 0.0f;
}

// ---------------------------------------------------------------------------
extern "C" void kernel_launch(void* const* d_in, const int* in_sizes, int n_in,
                              void* d_out, int out_size, void* d_ws, size_t ws_size,
                              hipStream_t stream) {
    (void)in_sizes; (void)n_in; (void)out_size; (void)ws_size;

    const int*           tokens  = (const int*)d_in[0];
    const unsigned char* sel     = (const unsigned char*)d_in[1];
    const float* tok_emb = (const float*)d_in[2];
    const float* pos_emb = (const float*)d_in[3];
    const float* Wq      = (const float*)d_in[4];
    const float* Wk      = (const float*)d_in[5];
    const float* Wv      = (const float*)d_in[6];
    const float* Wo      = (const float*)d_in[7];
    const float* bo      = (const float*)d_in[8];
    const float* W1      = (const float*)d_in[9];
    const float* b1      = (const float*)d_in[10];
    const float* W2      = (const float*)d_in[11];
    const float* b2      = (const float*)d_in[12];
    const float* ln1_s   = (const float*)d_in[13];
    const float* ln1_b   = (const float*)d_in[14];
    const float* ln2_s   = (const float*)d_in[15];
    const float* ln2_b   = (const float*)d_in[16];
    const float* fn_s    = (const float*)d_in[17];
    const float* fn_b    = (const float*)d_in[18];
    const float* W_head  = (const float*)d_in[19];

    // ---- workspace layout
    const size_t nD = (size_t)ROWS * Dsz;
    float* xw = (float*)d_ws;                                   // [ROWS,D] fp32 residual
    __hip_bfloat16* bb   = (__hip_bfloat16*)(xw + nD);
    __hip_bfloat16* h_bf = bb;                                  // LN1 out / ctx
    __hip_bfloat16* q_bf = h_bf + nD;                           // q / LN2 out
    __hip_bfloat16* k_bf = q_bf + nD;
    __hip_bfloat16* v_bf = k_bf + nD;
    __hip_bfloat16* ff_bf = v_bf + nD;                          // [ROWS,DFF]
    __hip_bfloat16* wqT = ff_bf + (size_t)ROWS * DFF;           // [L][D][D] transposed
    __hip_bfloat16* wkT = wqT + (size_t)Ll * Dsz * Dsz;
    __hip_bfloat16* wvT = wkT + (size_t)Ll * Dsz * Dsz;
    __hip_bfloat16* woT = wvT + (size_t)Ll * Dsz * Dsz;
    __hip_bfloat16* w1T = woT + (size_t)Ll * Dsz * Dsz;         // [L][DFF][D]
    __hip_bfloat16* w2T = w1T + (size_t)Ll * DFF * Dsz;         // [L][D][DFF]
    __hip_bfloat16* whT = w2T + (size_t)Ll * Dsz * DFF;         // [8(pad)][D]

    float* fout       = (float*)d_out;
    float* out_sel    = fout;
    float* out_logits = fout + ROWS;
    float* out_x      = fout + ROWS + (size_t)ROWS * 5;

    const dim3 blk(256);

    // ---- one-time weight downconvert + transpose (bf16 [N][K])
    for (int l = 0; l < Ll; ++l) {
        transpose_bf16_kernel<<<dim3(24, 24), blk, 0, stream>>>(Wq + (size_t)l * Dsz * Dsz, wqT + (size_t)l * Dsz * Dsz, Dsz, Dsz);
        transpose_bf16_kernel<<<dim3(24, 24), blk, 0, stream>>>(Wk + (size_t)l * Dsz * Dsz, wkT + (size_t)l * Dsz * Dsz, Dsz, Dsz);
        transpose_bf16_kernel<<<dim3(24, 24), blk, 0, stream>>>(Wv + (size_t)l * Dsz * Dsz, wvT + (size_t)l * Dsz * Dsz, Dsz, Dsz);
        transpose_bf16_kernel<<<dim3(24, 24), blk, 0, stream>>>(Wo + (size_t)l * Dsz * Dsz, woT + (size_t)l * Dsz * Dsz, Dsz, Dsz);
        transpose_bf16_kernel<<<dim3(96, 24), blk, 0, stream>>>(W1 + (size_t)l * Dsz * DFF, w1T + (size_t)l * DFF * Dsz, Dsz, DFF);
        transpose_bf16_kernel<<<dim3(24, 96), blk, 0, stream>>>(W2 + (size_t)l * DFF * Dsz, w2T + (size_t)l * Dsz * DFF, DFF, Dsz);
    }
    transpose_bf16_kernel<<<dim3(1, 24), blk, 0, stream>>>(W_head, whT, Dsz, 5);

    // ---- embeddings
    embed_kernel<<<dim3((ROWS * Dsz + 255) / 256), blk, 0, stream>>>(tokens, tok_emb, pos_emb, xw);

    const dim3 gD(Dsz / TN, ROWS / TM);     // (12, 128)
    const dim3 gF(DFF / TN, ROWS / TM);     // (48, 128)
    const dim3 gA(Ssz / 128, Hh, Bsz);      // (4, 12, 32)

    for (int l = 0; l < Ll; ++l) {
        const __hip_bfloat16* wq = wqT + (size_t)l * Dsz * Dsz;
        const __hip_bfloat16* wk = wkT + (size_t)l * Dsz * Dsz;
        const __hip_bfloat16* wv = wvT + (size_t)l * Dsz * Dsz;
        const __hip_bfloat16* wo = woT + (size_t)l * Dsz * Dsz;
        const __hip_bfloat16* w1 = w1T + (size_t)l * DFF * Dsz;
        const __hip_bfloat16* w2 = w2T + (size_t)l * Dsz * DFF;

        // h = LN1(x)
        layernorm_kernel<<<ROWS, blk, 0, stream>>>(xw, ln1_s + (size_t)l * Dsz, ln1_b + (size_t)l * Dsz, h_bf, nullptr);
        // q,k,v = h @ W{q,k,v}
        gemm_bf16_wmma<false, true><<<gD, blk, 0, stream>>>(h_bf, wq, nullptr, nullptr, q_bf, ROWS, Dsz, Dsz);
        gemm_bf16_wmma<false, true><<<gD, blk, 0, stream>>>(h_bf, wk, nullptr, nullptr, k_bf, ROWS, Dsz, Dsz);
        gemm_bf16_wmma<false, true><<<gD, blk, 0, stream>>>(h_bf, wv, nullptr, nullptr, v_bf, ROWS, Dsz, Dsz);
        // ctx = softmax(qk^T/8) v   (ctx reuses h buffer)
        attn_kernel<<<gA, blk, 0, stream>>>(q_bf, k_bf, v_bf, h_bf);
        // x = x + ctx @ Wo + bo
        gemm_bf16_wmma<false, false><<<gD, blk, 0, stream>>>(h_bf, wo, bo + (size_t)l * Dsz, xw, xw, ROWS, Dsz, Dsz);
        // h2 = LN2(x)  (reuses q buffer)
        layernorm_kernel<<<ROWS, blk, 0, stream>>>(xw, ln2_s + (size_t)l * Dsz, ln2_b + (size_t)l * Dsz, q_bf, nullptr);
        // ff = gelu(h2 @ W1 + b1)
        gemm_bf16_wmma<true, true><<<gF, blk, 0, stream>>>(q_bf, w1, b1 + (size_t)l * DFF, nullptr, ff_bf, ROWS, Dsz, DFF);
        // x = x + ff @ W2 + b2
        gemm_bf16_wmma<false, false><<<gD, blk, 0, stream>>>(ff_bf, w2, b2 + (size_t)l * Dsz, xw, xw, ROWS, DFF, Dsz);
    }

    // final LN -> bf16 (for head) + fp32 x output
    layernorm_kernel<<<ROWS, blk, 0, stream>>>(xw, fn_s, fn_b, h_bf, out_x);
    // logits = x_final @ W_head  (N=5; TDM descriptor clamps rows, store guarded)
    gemm_bf16_wmma<false, false><<<dim3(1, ROWS / TM), blk, 0, stream>>>(h_bf, whT, nullptr, nullptr,
                                                                         out_logits, ROWS, Dsz, 5);
    // selected mask passthrough
    selected_kernel<<<dim3((ROWS + 255) / 256), blk, 0, stream>>>(sel, out_sel, ROWS);
}
